// GptOssAttention_75892072120973
// MI455X (gfx1250) — compile-verified, matrix operand order checked by reading, not run on hardware
//
#include <hip/hip_runtime.h>
#include <hip/hip_bf16.h>
#include <math.h>

// ---------------- problem constants ----------------
#define BB 4
#define LL 512
#define TT (BB * LL)          // 2048
#define DD 2880
#define NH 64                 // query heads
#define KH 8                  // kv heads
#define GG 8                  // heads per kv head
#define HH 64                 // head dim
#define WIN 128
#define SM_SCALE 0.125f       // 1/sqrt(64)
#define CQKV 5120             // fused q|k|v output columns (4096+512+512)

typedef __attribute__((ext_vector_type(16))) __bf16 v16bf;
typedef __attribute__((ext_vector_type(8)))  float  v8f;
typedef __attribute__((ext_vector_type(4)))  int    v4i;

#define ASG __attribute__((address_space(1)))   // global
#define ASL __attribute__((address_space(3)))   // LDS

#if defined(__has_builtin)
#if __has_builtin(__builtin_amdgcn_global_load_async_to_lds_b128)
#define USE_ASYNC_LDS 1
#endif
#if __has_builtin(__builtin_amdgcn_s_wait_asynccnt)
#define HAVE_WAIT_ASYNC_BUILTIN 1
#endif
#endif

// ---------------- helpers ----------------
__device__ __forceinline__ unsigned short f2bf(float f) {
    unsigned u = __float_as_uint(f);
    u += 0x7FFFu + ((u >> 16) & 1u);      // round-to-nearest-even
    return (unsigned short)(u >> 16);
}

// copy 16 bytes global -> LDS (async DMA on CDNA5; VGPR fallback otherwise)
__device__ __forceinline__ void stage16(const unsigned short* g,
                                        unsigned short* l) {
#ifdef USE_ASYNC_LDS
    // AS1: flat global address == global address.
    // AS3: low 32 bits of a flat LDS address are the LDS byte offset
    //      (ISA 10.2).
    ASG v4i* gp = (ASG v4i*)(unsigned long long)g;
    ASL v4i* lp = (ASL v4i*)(unsigned int)(unsigned long long)l;
    __builtin_amdgcn_global_load_async_to_lds_b128(gp, lp, 0, 0);
#else
    const uint4 v = *(const uint4*)g;
    *(uint4*)l = v;
#endif
}

__device__ __forceinline__ void stage_wait() {
#ifdef USE_ASYNC_LDS
#ifdef HAVE_WAIT_ASYNC_BUILTIN
    __builtin_amdgcn_s_wait_asynccnt(0);
#else
    asm volatile("s_wait_asynccnt 0" ::: "memory");
#endif
#endif
}

// 32B fragment from LDS via two 16B-aligned loads (row stride is 80B)
__device__ __forceinline__ v16bf load_frag(const unsigned short* p) {
    union { uint4 q[2]; v16bf v; } u;
    const uint4* p4 = reinterpret_cast<const uint4*>(p);
    u.q[0] = p4[0];
    u.q[1] = p4[1];
    return u.v;
}

// ------------- WMMA GEMM: out[M,C] = A[M,Kd] * Bt[C,Kd]^T + bias[C] --------
// A, Bt are bf16; Bt holds W transposed so both operands stream contiguously.
// BM=128, BN=256, BK=32; 256 threads = 8 waves; wave -> 64x64 (4x4 tiles).
// M, Kd multiples of 128/32; C guarded (Bt OOB rows zero-filled).
#define LSTR 40   // 32 halves + 8 pad -> 80B rows, keeps 16B alignment

__device__ __forceinline__ void stage_tile(
    const unsigned short* __restrict__ A, const unsigned short* __restrict__ Bt,
    unsigned short* As, unsigned short* Bs,
    int m0, int n0, int k0, int Kd, int C, int tid) {
    // A tile: 128 rows x 32 halves = 512 x 16B segments (2 per thread)
#pragma unroll
    for (int it = 0; it < 2; ++it) {
        const int s   = tid + it * 256;
        const int row = s >> 2;
        const int seg = (s & 3) << 3;
        stage16(A + (size_t)(m0 + row) * Kd + k0 + seg, &As[row * LSTR + seg]);
    }
    // Bt tile: 256 rows x 32 halves = 1024 x 16B segments (4 per thread)
#pragma unroll
    for (int it = 0; it < 4; ++it) {
        const int s   = tid + it * 256;
        const int row = s >> 2;
        const int seg = (s & 3) << 3;
        const int c   = n0 + row;
        unsigned short* l = &Bs[row * LSTR + seg];
        if (c < C) {
            stage16(Bt + (size_t)c * Kd + k0 + seg, l);
        } else {
            const uint4 z = {0u, 0u, 0u, 0u};
            *(uint4*)l = z;
        }
    }
}

__global__ __launch_bounds__(256)
void gemm_bf16_wmma(const unsigned short* __restrict__ A,
                    const unsigned short* __restrict__ Bt,
                    const float* __restrict__ bias,
                    float* __restrict__ out,
                    int M, int Kd, int C) {
    __shared__ unsigned short As[2][128 * LSTR];
    __shared__ unsigned short Bs[2][256 * LSTR];

    const int tid  = threadIdx.x;
    const int lane = tid & 31;
    const int wave = tid >> 5;
    const int m0 = blockIdx.x * 128;
    const int n0 = blockIdx.y * 256;
    const int wm = (wave & 1) * 64;
    const int wn = (wave >> 1) * 64;

    const v8f zero = {0.f, 0.f, 0.f, 0.f, 0.f, 0.f, 0.f, 0.f};
    v8f acc[4][4];
#pragma unroll
    for (int mi = 0; mi < 4; ++mi)
#pragma unroll
        for (int ni = 0; ni < 4; ++ni) acc[mi][ni] = zero;

    // prime first tile
    stage_tile(A, Bt, As[0], Bs[0], m0, n0, 0, Kd, C, tid);
    stage_wait();
    __syncthreads();

    int buf = 0;
    for (int k0 = 0; k0 < Kd; k0 += 32) {
        // stream next tile into the other buffer while computing this one
        if (k0 + 32 < Kd)
            stage_tile(A, Bt, As[buf ^ 1], Bs[buf ^ 1], m0, n0, k0 + 32, Kd, C,
                       tid);

        const int khalf = (lane >> 4) << 4;   // lane<16 -> K[0..15], else K[16..31]
        v16bf afr[4];
#pragma unroll
        for (int mi = 0; mi < 4; ++mi)
            afr[mi] =
                load_frag(&As[buf][(wm + mi * 16 + (lane & 15)) * LSTR + khalf]);
#pragma unroll
        for (int ni = 0; ni < 4; ++ni) {
            const v16bf bfr =
                load_frag(&Bs[buf][(wn + ni * 16 + (lane & 15)) * LSTR + khalf]);
#pragma unroll
            for (int mi = 0; mi < 4; ++mi)
                acc[mi][ni] = __builtin_amdgcn_wmma_f32_16x16x32_bf16(
                    false, afr[mi], false, bfr, (short)0, acc[mi][ni], false,
                    false);
        }
        stage_wait();
        __syncthreads();
        buf ^= 1;
    }

    // store: lane = N column; VGPR p = row p (+8 for lanes 16..31)
    const int rbase = m0 + wm + ((lane >> 4) << 3);
    const int cbase = n0 + wn + (lane & 15);
#pragma unroll
    for (int mi = 0; mi < 4; ++mi)
#pragma unroll
        for (int ni = 0; ni < 4; ++ni) {
            const int col = cbase + ni * 16;
            if (col < C) {
                const float bv = bias[col];
#pragma unroll
                for (int p = 0; p < 8; ++p)
                    out[(size_t)(rbase + mi * 16 + p) * C + col] =
                        acc[mi][ni][p] + bv;
            }
        }
}

// ------------- packing / conversion pre-passes (one-time, L2-resident) -----
// Wt[c][d] = {wq|wk|wv}[d][c']  -> fused transposed bf16 (5120 x 2880)
__global__ void pack_wqkv_t(const float* __restrict__ wq,
                            const float* __restrict__ wk,
                            const float* __restrict__ wv,
                            unsigned short* __restrict__ Wt) {
    const int idx = blockIdx.x * blockDim.x + threadIdx.x;
    if (idx >= CQKV * (DD / 4)) return;
    const int c  = idx / (DD / 4);
    const int d0 = (idx - c * (DD / 4)) * 4;
    const float* src;
    int stride, cc;
    if (c < 4096)      { src = wq; stride = 4096; cc = c; }
    else if (c < 4608) { src = wk; stride = 512;  cc = c - 4096; }
    else               { src = wv; stride = 512;  cc = c - 4608; }
    unsigned short* o = Wt + (size_t)c * DD + d0;
#pragma unroll
    for (int j = 0; j < 4; ++j)
        o[j] = f2bf(src[(size_t)(d0 + j) * stride + cc]);
}

__global__ void pack_bias_qkv(const float* __restrict__ bq,
                              const float* __restrict__ bk,
                              const float* __restrict__ bv,
                              float* __restrict__ biasq) {
    const int c = blockIdx.x * blockDim.x + threadIdx.x;
    if (c >= CQKV) return;
    biasq[c] = (c < 4096) ? bq[c] : (c < 4608) ? bk[c - 4096] : bv[c - 4608];
}

// Wt[c][k] = wo[k][c]  -> transposed bf16 (2880 x 4096)
__global__ void pack_wo_t(const float* __restrict__ wo,
                          unsigned short* __restrict__ Wt) {
    const int idx = blockIdx.x * blockDim.x + threadIdx.x;
    if (idx >= DD * 1024) return;
    const int c  = idx >> 10;
    const int k0 = (idx & 1023) * 4;
    unsigned short* o = Wt + (size_t)c * 4096 + k0;
#pragma unroll
    for (int j = 0; j < 4; ++j)
        o[j] = f2bf(wo[(size_t)(k0 + j) * DD + c]);
}

__global__ void convert_f32_bf16(const float* __restrict__ in,
                                 unsigned short* __restrict__ out, int n4) {
    const int i = blockIdx.x * blockDim.x + threadIdx.x;
    if (i >= n4) return;
    const float4 f = ((const float4*)in)[i];
    unsigned short* o = out + (size_t)i * 4;
    o[0] = f2bf(f.x); o[1] = f2bf(f.y); o[2] = f2bf(f.z); o[3] = f2bf(f.w);
}

// ---------------- RoPE tables (YaRN blend, exact reference math) -----------
__global__ void rope_tables(const int* __restrict__ positions,
                            float* __restrict__ cosb, float* __restrict__ sinb) {
    const int idx = blockIdx.x * blockDim.x + threadIdx.x;
    if (idx >= TT * 32) return;
    const int t = idx >> 5;
    const int j = idx & 31;
    const float theta  = 150000.0f;
    const float two_pi = 6.2831853071795864769f;
    const float freq = powf(theta, (float)j / 32.0f);
    const float concentration = 0.1f * logf(32.0f) + 1.0f;
    const float lnT  = logf(theta);
    const float low  = 32.0f * logf(4096.0f / (32.0f * two_pi)) / lnT;
    const float high = 32.0f * logf(4096.0f / (1.0f  * two_pi)) / lnT;
    const float interp = 1.0f / (32.0f * freq);
    const float extrap = 1.0f / freq;
    float ramp = ((float)j - low) / (high - low);
    ramp = fminf(fmaxf(ramp, 0.0f), 1.0f);
    const float inv_freq = interp * ramp + extrap * (1.0f - ramp);
    const float ang = (float)positions[t] * inv_freq;
    cosb[idx] = cosf(ang) * concentration;
    sinb[idx] = sinf(ang) * concentration;
}

// ---------------- in-place RoPE application on the fused qkv buffer --------
__global__ void rope_apply(float* __restrict__ x,
                           const float* __restrict__ cosb,
                           const float* __restrict__ sinb,
                           int rowstride, int headoff, int nheads, int total) {
    const int idx = blockIdx.x * blockDim.x + threadIdx.x;
    if (idx >= total) return;
    const int j  = idx & 31;
    const int tn = idx >> 5;
    const int n  = tn % nheads;
    const int t  = tn / nheads;
    float* base = x + (size_t)t * rowstride + headoff + n * HH;
    const float x1 = base[j];
    const float x2 = base[j + 32];
    const float c = cosb[t * 32 + j];
    const float s = sinb[t * 32 + j];
    base[j]      = x1 * c - x2 * s;
    base[j + 32] = x2 * c + x1 * s;
}

// ------ sliding-window attention with sinks (online softmax, bf16 out) -----
// grid = B*KH*(L/32) blocks of 256; thread = (g = tid/32, query = chunk*32+lane)
__global__ __launch_bounds__(256)
void attn_win_kernel(const float* __restrict__ qkv,
                     const float* __restrict__ sinks,
                     unsigned short* __restrict__ attnb) {
    const int blk   = blockIdx.x;
    const int chunk = blk & 15;          // L/32 = 16 chunks
    const int kh    = (blk >> 4) & 7;
    const int b     = blk >> 7;
    const int g     = threadIdx.x >> 5;
    const int lane  = threadIdx.x & 31;
    const int qi    = chunk * 32 + lane;
    const int n     = kh * GG + g;
    const int t     = b * LL + qi;

    const float* qp = qkv + (size_t)t * CQKV + n * HH;
    float qv[HH];
#pragma unroll
    for (int h = 0; h < HH; ++h) qv[h] = qp[h];

    float m = -1e30f, d = 0.0f;
    float ov[HH];
#pragma unroll
    for (int h = 0; h < HH; ++h) ov[h] = 0.0f;

    int j0 = qi - (WIN - 1);
    if (j0 < 0) j0 = 0;
    for (int j = j0; j <= qi; ++j) {
        const size_t rowbase = (size_t)(b * LL + j) * CQKV;
        const float4* kp4 = (const float4*)(qkv + rowbase + 4096 + kh * HH);
        float s = 0.0f;
#pragma unroll
        for (int h4 = 0; h4 < 16; ++h4) {
            const float4 kk = kp4[h4];
            s += qv[4 * h4 + 0] * kk.x + qv[4 * h4 + 1] * kk.y +
                 qv[4 * h4 + 2] * kk.z + qv[4 * h4 + 3] * kk.w;
        }
        s *= SM_SCALE;
        const float nm   = fmaxf(m, s);
        const float corr = __expf(m - nm);
        const float p    = __expf(s - nm);
        d = d * corr + p;
        const float4* vp4 = (const float4*)(qkv + rowbase + 4608 + kh * HH);
#pragma unroll
        for (int h4 = 0; h4 < 16; ++h4) {
            const float4 vv = vp4[h4];
            ov[4 * h4 + 0] = ov[4 * h4 + 0] * corr + p * vv.x;
            ov[4 * h4 + 1] = ov[4 * h4 + 1] * corr + p * vv.y;
            ov[4 * h4 + 2] = ov[4 * h4 + 2] * corr + p * vv.z;
            ov[4 * h4 + 3] = ov[4 * h4 + 3] * corr + p * vv.w;
        }
        m = nm;
    }
    // fold in the attention sink
    const float sink = sinks[n];
    const float nm   = fmaxf(m, sink);
    const float corr = __expf(m - nm);
    d = d * corr + __expf(sink - nm);
    const float scale = corr / d;

    // write bf16 directly: this is the A operand of the output GEMM
    unsigned short* op = attnb + (size_t)t * 4096 + n * HH;
#pragma unroll
    for (int h = 0; h < HH; ++h) op[h] = f2bf(ov[h] * scale);
}

// ---------------- host-side orchestration ----------------
extern "C" void kernel_launch(void* const* d_in, const int* in_sizes, int n_in,
                              void* d_out, int out_size, void* d_ws,
                              size_t ws_size, hipStream_t stream) {
    (void)in_sizes; (void)n_in; (void)out_size; (void)ws_size;
    const float* x     = (const float*)d_in[0];
    const float* wq    = (const float*)d_in[1];
    const float* bq    = (const float*)d_in[2];
    const float* wk    = (const float*)d_in[3];
    const float* bk    = (const float*)d_in[4];
    const float* wv    = (const float*)d_in[5];
    const float* bv    = (const float*)d_in[6];
    const float* wo    = (const float*)d_in[7];
    const float* bo    = (const float*)d_in[8];
    const float* sinks = (const float*)d_in[9];
    const int*   pos   = (const int*)d_in[10];
    float* out = (float*)d_out;

    // workspace layout (~89 MB)
    char* w = (char*)d_ws;
    float* qkv = (float*)w;           w += (size_t)TT * CQKV * 4;  // 40 MB f32
    float* biasq = (float*)w;         w += (size_t)CQKV * 4;
    float* cosb = (float*)w;          w += (size_t)TT * 32 * 4;
    float* sinb = (float*)w;          w += (size_t)TT * 32 * 4;
    unsigned short* Wt = (unsigned short*)w;                        // reused
    w += (size_t)CQKV * DD * 2;                                     // 29.5 MB
    unsigned short* Ab = (unsigned short*)w;                        // reused
    w += (size_t)TT * 4096 * 2;                                     // 16.8 MB

    const dim3 blk(256);

    // pre-passes: pack transposed bf16 weights, fused bias, bf16 x, RoPE tables
    pack_wqkv_t<<<(CQKV * (DD / 4) + 255) / 256, blk, 0, stream>>>(wq, wk, wv,
                                                                   Wt);
    pack_bias_qkv<<<(CQKV + 255) / 256, blk, 0, stream>>>(bq, bk, bv, biasq);
    convert_f32_bf16<<<((TT * DD / 4) + 255) / 256, blk, 0, stream>>>(
        x, Ab, TT * DD / 4);
    rope_tables<<<(TT * 32 + 255) / 256, blk, 0, stream>>>(pos, cosb, sinb);

    // fused QKV projection: qkv(T,5120) = x * [wq|wk|wv] + bias
    gemm_bf16_wmma<<<dim3(TT / 128, CQKV / 256), blk, 0, stream>>>(
        Ab, Wt, biasq, qkv, TT, DD, CQKV);

    // RoPE on q (cols 0..4095) and k (cols 4096..4607), in place
    rope_apply<<<(TT * NH * 32 + 255) / 256, blk, 0, stream>>>(
        qkv, cosb, sinb, CQKV, 0, NH, TT * NH * 32);
    rope_apply<<<(TT * KH * 32 + 255) / 256, blk, 0, stream>>>(
        qkv, cosb, sinb, CQKV, 4096, KH, TT * KH * 32);

    // windowed attention with sinks -> bf16 activations (reuses Ab)
    attn_win_kernel<<<BB * KH * (LL / 32), blk, 0, stream>>>(qkv, sinks, Ab);

    // output projection: out(T,2880) = attn * wo + bo  (Wt reused for wo^T)
    pack_wo_t<<<(DD * 1024 + 255) / 256, blk, 0, stream>>>(wo, Wt);
    gemm_bf16_wmma<<<dim3(TT / 128, (DD + 255) / 256), blk, 0, stream>>>(
        Ab, Wt, bo, out, TT, NH * HH, DD);
}
